// MemristorLinearLayer_14525579395745
// MI455X (gfx1250) — matrix operand order repaired and checked by reading, not compiled, not run on hardware
//
#include <hip/hip_runtime.h>

// MemristorLinearLayer == out = x @ W^T + bias (the G_pos/G_neg construction
// cancels exactly: V@(G_pos-G_neg)/k_I = inputs@wb). Implemented as an f16
// split GEMM (hi/lo halves, 3 WMMA products) so precision ~fp32 while using
// the CDNA5 v_wmma_f32_16x16x32_f16 matrix pipe.

typedef __attribute__((ext_vector_type(16))) _Float16 v16h;
typedef __attribute__((ext_vector_type(8)))  _Float16 v8h;
typedef __attribute__((ext_vector_type(4)))  _Float16 v4h;
typedef __attribute__((ext_vector_type(8)))  float    v8f;
typedef __attribute__((ext_vector_type(4)))  float    v4f;

constexpr int M = 512, N = 2048, K = 2048;
constexpr int BM = 64, BN = 128, BK = 32;
constexpr int LD = 40;            // padded LDS row stride (f16 elems): 20 dwords -> conflict-free b128 frag loads
constexpr int KSTEPS = K / BK;    // 64

// A fragment (16x32 f16, per ISA 7.12.2): lane row = l&15,
// elems 0..7 = K kb..kb+7, elems 8..15 = K kb+16..kb+23, kb = (l>>4)*8.
__device__ __forceinline__ v16h load_frag_a(const _Float16* s, int row, int kb) {
  const _Float16* p = s + row * LD + kb;
  v8h x0 = *(const v8h*)(p);
  v8h x1 = *(const v8h*)(p + 16);
  return __builtin_shufflevector(x0, x1, 0,1,2,3,4,5,6,7,8,9,10,11,12,13,14,15);
}

// B fragment (32x16 f16): lane col = l&15, elems 0..7 = K kh..kh+7, elems 8..15 = K kh+16..kh+23, kh = (l>>4)*8.
// B = W^T, so B columns are W rows -> same row-major access as A.
__device__ __forceinline__ v16h load_frag_b(const _Float16* s, int row, int kh) {
  const _Float16* p = s + row * LD + kh;
  v8h x0 = *(const v8h*)(p);
  v8h x1 = *(const v8h*)(p + 16);
  return __builtin_shufflevector(x0, x1, 0,1,2,3,4,5,6,7,8,9,10,11,12,13,14,15);
}

__device__ __forceinline__ void split_store(_Float16* sh, _Float16* sl, int off, v4f v) {
  v4h hi, lo;
#pragma unroll
  for (int e = 0; e < 4; ++e) {
    _Float16 h = (_Float16)v[e];
    hi[e] = h;
    lo[e] = (_Float16)(v[e] - (float)h);   // exact residual, fits f16 (denorms preserved by WMMA)
  }
  *(v4h*)(sh + off) = hi;
  *(v4h*)(sl + off) = lo;
}

__global__ __launch_bounds__(256) void memristor_linear_wmma(
    const float* __restrict__ x, const float* __restrict__ w,
    const float* __restrict__ bias, float* __restrict__ out)
{
  __shared__ __align__(16) _Float16 sAh[BM * LD];
  __shared__ __align__(16) _Float16 sAl[BM * LD];
  __shared__ __align__(16) _Float16 sBh[BN * LD];
  __shared__ __align__(16) _Float16 sBl[BN * LD];

  const int tid  = threadIdx.x;
  const int lane = tid & 31;
  const int wave = tid >> 5;          // 8 waves: 2 (M) x 4 (N)
  const int m0 = blockIdx.y * BM;
  const int n0 = blockIdx.x * BN;
  const int wm = (wave >> 2) * 32;    // wave patch origin in block tile
  const int wn = (wave & 3)  * 32;

  // Global staging: A tile 64x32 f32 (2 float4/thread), B tile 128x32 f32 (4 float4/thread)
  v4f ga[2], gb[4];
  auto gload = [&](int t) {
    const int k0 = t * BK;
#pragma unroll
    for (int j = 0; j < 2; ++j) {
      const int idx = tid + 256 * j;
      const int r = idx >> 3, c = idx & 7;
      ga[j] = *(const v4f*)(x + (size_t)(m0 + r) * K + k0 + c * 4);
    }
#pragma unroll
    for (int j = 0; j < 4; ++j) {
      const int idx = tid + 256 * j;
      const int r = idx >> 3, c = idx & 7;
      gb[j] = *(const v4f*)(w + (size_t)(n0 + r) * K + k0 + c * 4);
    }
  };

  v8f acc[2][2] = {};
  const int fr = lane & 15;
  const int ka = (lane >> 4) << 3;    // A K-base: 0 or 8
  const int kb = (lane >> 4) << 4;    // unused (0 or 16); B fragments also use ka

  gload(0);
  for (int t = 0; t < KSTEPS; ++t) {
    __syncthreads();                  // previous step done reading LDS
#pragma unroll
    for (int j = 0; j < 2; ++j) {
      const int idx = tid + 256 * j;
      split_store(sAh, sAl, (idx >> 3) * LD + (idx & 7) * 4, ga[j]);
    }
#pragma unroll
    for (int j = 0; j < 4; ++j) {
      const int idx = tid + 256 * j;
      split_store(sBh, sBl, (idx >> 3) * LD + (idx & 7) * 4, gb[j]);
    }
    __syncthreads();

    if (t + 1 < KSTEPS) gload(t + 1);  // hide HBM latency under the WMMA chain

    v16h ah[2], al[2], bh[2], bl[2];
#pragma unroll
    for (int i = 0; i < 2; ++i) {
      ah[i] = load_frag_a(sAh, wm + i * 16 + fr, ka);
      al[i] = load_frag_a(sAl, wm + i * 16 + fr, ka);
    }
#pragma unroll
    for (int j = 0; j < 2; ++j) {
      bh[j] = load_frag_b(sBh, wn + j * 16 + fr, ka);
      bl[j] = load_frag_b(sBl, wn + j * 16 + fr, ka);
    }
#pragma unroll
    for (int i = 0; i < 2; ++i)
#pragma unroll
      for (int j = 0; j < 2; ++j) {
        acc[i][j] = __builtin_amdgcn_wmma_f32_16x16x32_f16(
            false, ah[i], false, bh[j], (short)0, acc[i][j], false, false);
        acc[i][j] = __builtin_amdgcn_wmma_f32_16x16x32_f16(
            false, ah[i], false, bl[j], (short)0, acc[i][j], false, false);
        acc[i][j] = __builtin_amdgcn_wmma_f32_16x16x32_f16(
            false, al[i], false, bh[j], (short)0, acc[i][j], false, false);
      }
  }

  // Epilogue: C/D layout — lanes 0-15: N=lane, M=r; lanes 16-31: N=lane-16, M=8+r
  const int colBase = n0 + wn + (lane & 15);
  const int rowBase = m0 + wm + ((lane >> 4) << 3);
#pragma unroll
  for (int j = 0; j < 2; ++j) {
    const int col = colBase + j * 16;
    const float bj = bias[col];
#pragma unroll
    for (int i = 0; i < 2; ++i)
#pragma unroll
      for (int r = 0; r < 8; ++r)
        out[(size_t)(rowBase + i * 16 + r) * N + col] = acc[i][j][r] + bj;
  }
}

extern "C" void kernel_launch(void* const* d_in, const int* in_sizes, int n_in,
                              void* d_out, int out_size, void* d_ws, size_t ws_size,
                              hipStream_t stream) {
  (void)in_sizes; (void)n_in; (void)out_size; (void)d_ws; (void)ws_size;
  const float* x    = (const float*)d_in[0];   // (512, 2048)
  const float* w    = (const float*)d_in[1];   // (2048, 2048), row-major (out, in)
  const float* bias = (const float*)d_in[2];   // (2048,)
  float* out        = (float*)d_out;           // (512, 2048)
  dim3 grid(N / BN, M / BM);                   // (16, 8) = 128 workgroups
  hipLaunchKernelGGL(memristor_linear_wmma, grid, dim3(256), 0, stream, x, w, bias, out);
}
